// Mamba2_36481452212241
// MI455X (gfx1250) — matrix operand order repaired and checked by reading, not compiled, not run on hardware
//
#include <hip/hip_runtime.h>
#include <hip/hip_bf16.h>

// ---------------------------------------------------------------------------
// Mamba2 forward for MI455X (gfx1250), bf16 WMMA everywhere GEMM-shaped.
// B=2, L=2048, D_MODEL=2048, D_INNER=4096, H=64 heads x 64, D_STATE=128,
// G=8 groups, CHUNK=256, nc=8, D_IN_PROJ=10304, CONV_DIM=6144.
// ---------------------------------------------------------------------------

#define NB 2
#define NL 2048
#define DMODEL 2048
#define DINNER 4096
#define HEADS 64
#define HDIM 64
#define DSTATE 128
#define NGRP 8
#define NCHUNK 8
#define CHUNKL 256
#define DPROJ 10304
#define CONVD 6144
#define DTOFF 10240   // D_IN_PROJ - NHEADS

typedef __bf16 bf16_t;
typedef __attribute__((ext_vector_type(16))) __bf16 v16bf;
typedef __attribute__((ext_vector_type(8)))  float  v8f;

union Frag16 { v16bf v; unsigned short u[16]; uint4 q[2]; };

static __device__ __forceinline__ unsigned short f2bf(float f) {
  unsigned int u = __float_as_uint(f);
  u = u + 0x7FFFu + ((u >> 16) & 1u);   // round-to-nearest-even
  return (unsigned short)(u >> 16);
}
static __device__ __forceinline__ float bf2f(unsigned short h) {
  return __uint_as_float(((unsigned int)h) << 16);
}
static __device__ __forceinline__ v8f vzero() {
  v8f z;
  #pragma unroll
  for (int i = 0; i < 8; ++i) z[i] = 0.f;
  return z;
}
#define WMMA_BF16(A, B, C) \
  __builtin_amdgcn_wmma_f32_16x16x32_bf16(false, (A), false, (B), (short)0, (C), false, false)

// ------------------------------ fp32 -> bf16 -------------------------------
__global__ void k_cvt_bf16(const float* __restrict__ in, unsigned short* __restrict__ out, int n) {
  int i = blockIdx.x * 256 + threadIdx.x;
  if (i < n) out[i] = f2bf(in[i]);
}

// ------------------------- GEMM: C[M,N] = A[M,K] * B[N,K]^T ----------------
// A,B bf16 row-major (K contiguous), C fp32. Block = 64 thr (2 waves),
// wave tile 64M x 32N (4x2 frags), K step 32, straight-line prefetch.
__global__ void k_gemm_bf16(const unsigned short* __restrict__ A,
                            const unsigned short* __restrict__ Bm,
                            float* __restrict__ C, int M, int N, int K) {
  const int lane = threadIdx.x & 31;
  const int wave = threadIdx.x >> 5;
  const int l16  = lane & 15;
  const int half = lane >> 4;
  const int nbase = blockIdx.x * 64 + wave * 32;
  const int mbase = blockIdx.y * 64;

  const unsigned short* arow[4];
  const unsigned short* brow[2];
  #pragma unroll
  for (int i = 0; i < 4; ++i)
    arow[i] = A + (size_t)(mbase + i * 16 + l16) * K + half * 8;
  #pragma unroll
  for (int j = 0; j < 2; ++j)
    brow[j] = Bm + (size_t)(nbase + j * 16 + l16) * K + half * 8;

  v8f acc[4][2];
  #pragma unroll
  for (int i = 0; i < 4; ++i)
    #pragma unroll
    for (int j = 0; j < 2; ++j) acc[i][j] = vzero();

  for (int k0 = 0; k0 < K; k0 += 32) {
    Frag16 a[4];
    #pragma unroll
    for (int i = 0; i < 4; ++i) {
      const unsigned short* p = arow[i] + k0;
      a[i].q[0] = *(const uint4*)p;
      a[i].q[1] = *(const uint4*)(p + 16);
      __builtin_prefetch((const void*)(p + 32), 0, 1);   // speculative: OOB dropped
    }
    Frag16 b[2];
    #pragma unroll
    for (int j = 0; j < 2; ++j) {
      const unsigned short* p = brow[j] + k0;
      b[j].q[0] = *(const uint4*)p;
      b[j].q[1] = *(const uint4*)(p + 16);
      __builtin_prefetch((const void*)(p + 32), 0, 1);
    }
    #pragma unroll
    for (int i = 0; i < 4; ++i)
      #pragma unroll
      for (int j = 0; j < 2; ++j)
        acc[i][j] = WMMA_BF16(a[i].v, b[j].v, acc[i][j]);
  }
  #pragma unroll
  for (int i = 0; i < 4; ++i)
    #pragma unroll
    for (int j = 0; j < 2; ++j) {
      int n = nbase + j * 16 + l16;
      #pragma unroll
      for (int r = 0; r < 8; ++r) {
        int m = mbase + i * 16 + half * 8 + r;
        C[(size_t)m * N + n] = acc[i][j][r];
      }
    }
}

// ------------------------------- dt softplus -------------------------------
__global__ void k_dt(const float* __restrict__ zx, const float* __restrict__ dt_bias,
                     float* __restrict__ dtbuf) {
  int i = blockIdx.x * 256 + threadIdx.x;            // B*L*H = 262144
  if (i >= NB * NL * HEADS) return;
  int h = i & 63, bl = i >> 6;
  float x = zx[(size_t)bl * DPROJ + DTOFF + h] + dt_bias[h];
  dtbuf[i] = (x > 20.f) ? x : log1pf(expf(x));
}

// --------------------- per-chunk cumsum of A*dt (scan) ---------------------
// blockIdx.x = ((b*64+h)*8 + ch), 256 threads.
__global__ void k_acs(const float* __restrict__ dtbuf, const float* __restrict__ Ap,
                      float* __restrict__ acs, float* __restrict__ alast) {
  int ch = blockIdx.x & 7, bh = blockIdx.x >> 3;
  int h = bh & 63, b = bh >> 6;
  int s = threadIdx.x;
  int l = ch * CHUNKL + s;
  float a = dtbuf[((size_t)(b * NL + l)) * HEADS + h] * Ap[h];
  __shared__ float sm[CHUNKL];
  sm[s] = a; __syncthreads();
  for (int off = 1; off < CHUNKL; off <<= 1) {
    float v = (s >= off) ? sm[s - off] : 0.f;
    __syncthreads();
    sm[s] += v;
    __syncthreads();
  }
  float cs = sm[s];
  acs[(size_t)blockIdx.x * CHUNKL + s] = cs;
  if (s == 255) alast[blockIdx.x] = cs;
}

// -------------- conv1d(4) + SiLU + bf16 restaging for the SSD --------------
// grid (CONVD/256, B*L); writes convx fp32, B[s,n], B^T[n,s], C[s,n], (x*dt)^T[p,s]
__global__ void k_conv(const float* __restrict__ zx, const float* __restrict__ cw,
                       const float* __restrict__ cb, const float* __restrict__ dtbuf,
                       float* __restrict__ convx,
                       unsigned short* __restrict__ Bb, unsigned short* __restrict__ BbT,
                       unsigned short* __restrict__ Cbf, unsigned short* __restrict__ xdtT) {
  int c  = blockIdx.x * 256 + threadIdx.x;
  int bl = blockIdx.y;
  int b = bl >> 11, l = bl & 2047;
  float acc = cb[c];
  #pragma unroll
  for (int k = 0; k < 4; ++k) {
    int ll = l - 3 + k;
    if (ll >= 0) acc += cw[c * 4 + k] * zx[((size_t)(b * NL + ll)) * DPROJ + DINNER + c];
  }
  float s = acc / (1.f + expf(-acc));                 // SiLU
  int ch = l >> 8, si = l & 255;
  int bch = b * NCHUNK + ch;
  if (c < DINNER) {
    convx[(size_t)bl * DINNER + c] = s;
    int h = c >> 6, p = c & 63;
    float dtv = dtbuf[(size_t)bl * HEADS + h];
    xdtT[(((size_t)bch * HEADS + h) * HDIM + p) * CHUNKL + si] = f2bf(s * dtv);
  } else if (c < DINNER + NGRP * DSTATE) {
    int g = (c - DINNER) >> 7, n = (c - DINNER) & 127;
    size_t base = (size_t)bch * NGRP + g;
    unsigned short v = f2bf(s);
    Bb [(base * CHUNKL + si) * DSTATE + n] = v;
    BbT[(base * DSTATE + n) * CHUNKL + si] = v;
  } else {
    int g = (c - DINNER - NGRP * DSTATE) >> 7, n = (c - DINNER - NGRP * DSTATE) & 127;
    size_t base = (size_t)bch * NGRP + g;
    Cbf[(base * CHUNKL + si) * DSTATE + n] = f2bf(s);
  }
}

// ------- states[p,n] = sum_s exp(Alast-Acs[s]) * xdt[s,p] * B[s,n] ---------
// blockIdx.x = (b*8+ch)*64 + h; 8 waves: wave -> (p-tile = w&3, n-half = w>>2)
__global__ void k_states(const unsigned short* __restrict__ xdtT,
                         const unsigned short* __restrict__ BbT,
                         const float* __restrict__ acs, const float* __restrict__ alast,
                         float* __restrict__ states) {
  int h = blockIdx.x & 63, bch = blockIdx.x >> 6;
  int ch = bch & 7, b = bch >> 3;
  int g = h >> 3;
  __shared__ float ws[CHUNKL];
  {
    int s = threadIdx.x;
    size_t ai = (((size_t)(b * HEADS + h) * NCHUNK) + ch) * CHUNKL + s;
    float al = alast[(b * HEADS + h) * NCHUNK + ch];
    ws[s] = expf(al - acs[ai]);
  }
  __syncthreads();
  int lane = threadIdx.x & 31, wave = threadIdx.x >> 5;
  int l16 = lane & 15, half = lane >> 4;
  int pt = wave & 3, nh = wave >> 2;
  const unsigned short* xrow  = xdtT + (((size_t)bch * HEADS + h) * HDIM + pt * 16 + l16) * CHUNKL;
  const unsigned short* Bbase = BbT + ((size_t)bch * NGRP + g) * DSTATE * CHUNKL;

  v8f acc[4];
  #pragma unroll
  for (int j = 0; j < 4; ++j) acc[j] = vzero();

  for (int k0 = 0; k0 < CHUNKL; k0 += 32) {
    Frag16 a;
    int r0 = k0 + half * 8, r1 = k0 + 16 + half * 8;
    #pragma unroll
    for (int t = 0; t < 8; ++t) {
      a.u[t]     = f2bf(bf2f(xrow[r0 + t]) * ws[r0 + t]);
      a.u[8 + t] = f2bf(bf2f(xrow[r1 + t]) * ws[r1 + t]);
    }
    #pragma unroll
    for (int j = 0; j < 4; ++j) {
      const unsigned short* p = Bbase + (size_t)(nh * 64 + j * 16 + l16) * CHUNKL + k0 + half * 8;
      Frag16 bf; bf.q[0] = *(const uint4*)p; bf.q[1] = *(const uint4*)(p + 16);
      acc[j] = WMMA_BF16(a.v, bf.v, acc[j]);
    }
  }
  size_t sbase = (size_t)blockIdx.x * HDIM * DSTATE;
  #pragma unroll
  for (int j = 0; j < 4; ++j)
    #pragma unroll
    for (int r = 0; r < 8; ++r)
      states[sbase + (size_t)(pt * 16 + half * 8 + r) * DSTATE + nh * 64 + j * 16 + l16] = acc[j][r];
}

// ------------- inter-chunk recurrence: prev[z]=prev[z-1]*e+states ----------
// blockIdx.x = b*64+h; 256 threads x 32 strided elements of the 64x128 state.
__global__ void k_chunkscan(const float* __restrict__ states, const float* __restrict__ alast,
                            unsigned short* __restrict__ prevbf) {
  int bh = blockIdx.x;
  int h = bh & 63, b = bh >> 6;
  float cur[32];
  #pragma unroll
  for (int i = 0; i < 32; ++i) cur[i] = 0.f;
  for (int z = 0; z < NCHUNK; ++z) {
    size_t base = (((size_t)(b * NCHUNK + z) * HEADS) + h) * (HDIM * DSTATE);
    float dec = expf(alast[bh * NCHUNK + z]);
    #pragma unroll
    for (int i = 0; i < 32; ++i) {
      int e = threadIdx.x + 256 * i;
      prevbf[base + e] = f2bf(cur[i]);
      cur[i] = cur[i] * dec + states[base + e];
    }
  }
}

// --------------------- Y = Y_diag + Y_off  (per b,chunk,head) --------------
// blockIdx.x = (b*8+ch)*64 + h; 8 waves, wave handles l-tiles {w, 15-w}.
__global__ void k_y(const unsigned short* __restrict__ Bb, const unsigned short* __restrict__ Cbf,
                    const unsigned short* __restrict__ xdtT, const unsigned short* __restrict__ prevbf,
                    const float* __restrict__ acs, float* __restrict__ ybuf) {
  int h = blockIdx.x & 63, bch = blockIdx.x >> 6;
  int ch = bch & 7, b = bch >> 3;
  int g = h >> 3;
  __shared__ float sacs[CHUNKL];
  __shared__ __align__(16) unsigned short stage[8][16 * 32];
  sacs[threadIdx.x] = acs[((((size_t)(b * HEADS + h) * NCHUNK) + ch) * CHUNKL) + threadIdx.x];
  __syncthreads();

  int lane = threadIdx.x & 31, wave = threadIdx.x >> 5;
  int l16 = lane & 15, half = lane >> 4;
  const unsigned short* Crows = Cbf + ((size_t)bch * NGRP + g) * CHUNKL * DSTATE;
  const unsigned short* Brows = Bb  + ((size_t)bch * NGRP + g) * CHUNKL * DSTATE;
  const unsigned short* prows = prevbf + (size_t)blockIdx.x * (HDIM * DSTATE);
  const unsigned short* xrows = xdtT + (size_t)blockIdx.x * (HDIM * CHUNKL);

  for (int hi = 0; hi < 2; ++hi) {
    int lt = hi ? (15 - wave) : wave;
    int lbase = lt * 16;

    // C row fragments (A operand, K over n=0..127)
    Frag16 cl[4];
    const unsigned short* crow = Crows + (size_t)(lbase + l16) * DSTATE;
    #pragma unroll
    for (int kk = 0; kk < 4; ++kk) {
      cl[kk].q[0] = *(const uint4*)(crow + kk * 32 + half * 8);
      cl[kk].q[1] = *(const uint4*)(crow + kk * 32 + 16 + half * 8);
    }

    // Y_off = C @ prev^T, row-scaled by exp(Acs[l]) afterwards
    v8f yacc[4];
    #pragma unroll
    for (int j = 0; j < 4; ++j) {
      v8f yo = vzero();
      #pragma unroll
      for (int kk = 0; kk < 4; ++kk) {
        const unsigned short* p = prows + (size_t)(j * 16 + l16) * DSTATE + kk * 32 + half * 8;
        Frag16 pf; pf.q[0] = *(const uint4*)p; pf.q[1] = *(const uint4*)(p + 16);
        yo = WMMA_BF16(cl[kk].v, pf.v, yo);
      }
      #pragma unroll
      for (int r = 0; r < 8; ++r)
        yacc[j][r] = yo[r] * expf(sacs[lbase + half * 8 + r]);
    }

    // Y_diag: loop pairs of s-tiles (K=32) up to the diagonal tile
    for (int stp = 0; 2 * stp <= lt; ++stp) {
      int st0 = 2 * stp;
      #pragma unroll
      for (int u = 0; u < 2; ++u) {
        int st = st0 + u;
        if (st <= lt) {
          v8f gacc = vzero();
          const unsigned short* brow = Brows + (size_t)(st * 16 + l16) * DSTATE;
          #pragma unroll
          for (int kk = 0; kk < 4; ++kk) {
            Frag16 bf;
            bf.q[0] = *(const uint4*)(brow + kk * 32 + half * 8);
            bf.q[1] = *(const uint4*)(brow + kk * 32 + 16 + half * 8);
            gacc = WMMA_BF16(cl[kk].v, bf.v, gacc);
          }
          int sg = st * 16 + l16;
          #pragma unroll
          for (int r = 0; r < 8; ++r) {
            int lg = lbase + half * 8 + r;
            float v = (lg >= sg) ? gacc[r] * expf(sacs[lg] - sacs[sg]) : 0.f;
            stage[wave][(half * 8 + r) * 32 + u * 16 + l16] = f2bf(v);
          }
        } else {
          #pragma unroll
          for (int r = 0; r < 8; ++r)
            stage[wave][(half * 8 + r) * 32 + u * 16 + l16] = 0;
        }
      }
      // cross-lane LDS transpose within the wave: drain DS pipe
      asm volatile("s_wait_dscnt 0" ::: "memory");
      Frag16 m;
      const unsigned short* srow = &stage[wave][l16 * 32];
      m.q[0] = *(const uint4*)(srow + half * 8);
      m.q[1] = *(const uint4*)(srow + 16 + half * 8);
      #pragma unroll
      for (int j = 0; j < 4; ++j) {
        const unsigned short* xp = xrows + (size_t)(j * 16 + l16) * CHUNKL + st0 * 16 + half * 8;
        Frag16 xf; xf.q[0] = *(const uint4*)xp; xf.q[1] = *(const uint4*)(xp + 16);
        yacc[j] = WMMA_BF16(m.v, xf.v, yacc[j]);
      }
    }

    // store Y fp32 as [bl][h*64+p]
    #pragma unroll
    for (int j = 0; j < 4; ++j)
      #pragma unroll
      for (int r = 0; r < 8; ++r) {
        int ltok = ch * CHUNKL + lbase + half * 8 + r;
        ybuf[(((size_t)(b * NL + ltok)) * HEADS + h) * HDIM + j * 16 + l16] = yacc[j][r];
      }
  }
}

// ------------- y = (Y + D*x) * silu(z); group(512) RMSNorm; bf16 -----------
__global__ void k_gate_norm(const float* __restrict__ ybuf, const float* __restrict__ convx,
                            const float* __restrict__ zx, const float* __restrict__ Dp,
                            const float* __restrict__ nw, unsigned short* __restrict__ ynorm) {
  int bl = blockIdx.x, t = threadIdx.x;
  int e0 = t * 16;
  const float* yrow = ybuf  + (size_t)bl * DINNER;
  const float* xrow = convx + (size_t)bl * DINNER;
  const float* zrow = zx    + (size_t)bl * DPROJ;
  float v[16]; float ss = 0.f;
  #pragma unroll
  for (int i = 0; i < 16; ++i) {
    int e = e0 + i;
    float y = yrow[e] + Dp[e >> 6] * xrow[e];
    float z = zrow[e];
    y *= z / (1.f + expf(-z));
    v[i] = y; ss += y * y;
  }
  // 32 lanes x 16 elems == one 512-wide norm group per wave (wave32)
  #pragma unroll
  for (int off = 16; off >= 1; off >>= 1) ss += __shfl_xor(ss, off, 32);
  float rstd = rsqrtf(ss * (1.f / 512.f) + 1e-5f);
  #pragma unroll
  for (int i = 0; i < 16; ++i)
    ynorm[(size_t)bl * DINNER + e0 + i] = f2bf(v[i] * rstd * nw[e0 + i]);
}

// ---------------------------------------------------------------------------
extern "C" void kernel_launch(void* const* d_in, const int* in_sizes, int n_in,
                              void* d_out, int out_size, void* d_ws, size_t ws_size,
                              hipStream_t stream) {
  const float* hs      = (const float*)d_in[0];
  const float* W_in    = (const float*)d_in[1];
  const float* conv_w  = (const float*)d_in[2];
  const float* conv_b  = (const float*)d_in[3];
  const float* Ap      = (const float*)d_in[4];
  const float* Dp      = (const float*)d_in[5];
  const float* dt_bias = (const float*)d_in[6];
  const float* nw      = (const float*)d_in[7];
  const float* W_out   = (const float*)d_in[8];
  float* out = (float*)d_out;
  char* ws = (char*)d_ws;

  size_t off = 0;
  auto take = [&](size_t bytes) { char* p = ws + off; off += (bytes + 255) & ~(size_t)255; return p; };
  unsigned short* hid_bf  = (unsigned short*)take((size_t)NB * NL * DMODEL * 2);
  unsigned short* win_bf  = (unsigned short*)take((size_t)DPROJ * DMODEL * 2);
  unsigned short* wout_bf = (unsigned short*)take((size_t)DMODEL * DINNER * 2);
  float*          zxb     = (float*)take((size_t)NB * NL * DPROJ * 4);
  float*          dtbuf   = (float*)take((size_t)NB * NL * HEADS * 4);
  float*          acs     = (float*)take((size_t)NB * HEADS * NL * 4);
  float*          alast   = (float*)take((size_t)NB * HEADS * NCHUNK * 4);
  float*          convx   = (float*)take((size_t)NB * NL * DINNER * 4);
  unsigned short* Bb      = (unsigned short*)take((size_t)NB * NCHUNK * NGRP * CHUNKL * DSTATE * 2);
  unsigned short* BbT     = (unsigned short*)take((size_t)NB * NCHUNK * NGRP * CHUNKL * DSTATE * 2);
  unsigned short* Cbf     = (unsigned short*)take((size_t)NB * NCHUNK * NGRP * CHUNKL * DSTATE * 2);
  unsigned short* xdtT    = (unsigned short*)take((size_t)NB * NCHUNK * HEADS * HDIM * CHUNKL * 2);
  float*          states  = (float*)take((size_t)NB * NCHUNK * HEADS * HDIM * DSTATE * 4);
  unsigned short* prevbf  = (unsigned short*)take((size_t)NB * NCHUNK * HEADS * HDIM * DSTATE * 2);
  float*          ybuf    = (float*)take((size_t)NB * NL * DINNER * 4);
  unsigned short* ynorm   = (unsigned short*)take((size_t)NB * NL * DINNER * 2);
  (void)ws_size; (void)in_sizes; (void)n_in; (void)out_size;

  int nh = NB * NL * DMODEL, nwi = DPROJ * DMODEL, nwo = DMODEL * DINNER;
  k_cvt_bf16<<<(nh  + 255) / 256, 256, 0, stream>>>(hs,    hid_bf,  nh);
  k_cvt_bf16<<<(nwi + 255) / 256, 256, 0, stream>>>(W_in,  win_bf,  nwi);
  k_cvt_bf16<<<(nwo + 255) / 256, 256, 0, stream>>>(W_out, wout_bf, nwo);

  // in_proj: [4096,10304] = hid[4096,2048] @ W_in[10304,2048]^T
  k_gemm_bf16<<<dim3(DPROJ / 64, (NB * NL) / 64), 64, 0, stream>>>(hid_bf, win_bf, zxb,
                                                                   NB * NL, DPROJ, DMODEL);
  k_dt<<<(NB * NL * HEADS) / 256, 256, 0, stream>>>(zxb, dt_bias, dtbuf);
  k_acs<<<NB * HEADS * NCHUNK, CHUNKL, 0, stream>>>(dtbuf, Ap, acs, alast);
  k_conv<<<dim3(CONVD / 256, NB * NL), 256, 0, stream>>>(zxb, conv_w, conv_b, dtbuf,
                                                         convx, Bb, BbT, Cbf, xdtT);
  k_states<<<NB * NCHUNK * HEADS, 256, 0, stream>>>(xdtT, BbT, acs, alast, states);
  k_chunkscan<<<NB * HEADS, 256, 0, stream>>>(states, alast, prevbf);
  k_y<<<NB * NCHUNK * HEADS, 256, 0, stream>>>(Bb, Cbf, xdtT, prevbf, acs, ybuf);
  k_gate_norm<<<NB * NL, 256, 0, stream>>>(ybuf, convx, zxb, Dp, nw, ynorm);

  // out_proj: [4096,2048] = ynorm[4096,4096] @ W_out[2048,4096]^T
  k_gemm_bf16<<<dim3(DMODEL / 64, (NB * NL) / 64), 64, 0, stream>>>(ynorm, wout_bf, out,
                                                                    NB * NL, DMODEL, DINNER);
}
